// AudioVectorQuantizer_43224550867115
// MI455X (gfx1250) — matrix operand order repaired and checked by reading, not compiled, not run on hardware
//
#include <hip/hip_runtime.h>

// ---- CDNA5 WMMA types ----
typedef __attribute__((ext_vector_type(16))) __bf16 v16bf;
typedef __attribute__((ext_vector_type(8)))  float  v8f;
typedef __attribute__((ext_vector_type(4)))  int    v4i;

#define DIMS   256
#define KCODES 1024
#define NROWS  65536            // B*N = 16*4096
#define ZQ_ELEMS (NROWS * DIMS)

// Optional CDNA5 async global->LDS path (DMA, tracked by ASYNCcnt)
#if defined(__has_builtin)
#if __has_builtin(__builtin_amdgcn_global_load_async_to_lds_b128)
#define HAVE_ASYNC_LDS 1
#endif
#endif

#ifndef HAVE_ASYNC_LDS
#define HAVE_ASYNC_LDS 0
#endif

#if HAVE_ASYNC_LDS
typedef __attribute__((address_space(1))) v4i gv4i;   // global int4
typedef __attribute__((address_space(3))) v4i lv4i;   // LDS int4

// global_load_async_to_lds_b128: same imm offset applies to both addresses.
// Offset must be an immediate -> non-type template parameter.
template <int OFF>
__device__ __forceinline__ void async_cp16(const void* g, void* l) {
  __builtin_amdgcn_global_load_async_to_lds_b128((gv4i*)g, (lv4i*)l, OFF, 0);
}

#if __has_builtin(__builtin_amdgcn_s_wait_asynccnt)
#define ASYNC_WAIT() __builtin_amdgcn_s_wait_asynccnt(0)
#else
#define ASYNC_WAIT() asm volatile("s_wait_asynccnt 0x0" ::: "memory")
#endif
#endif

// f32 -> bf16, round-to-nearest-even
__device__ __forceinline__ unsigned short f2bf(float f) {
  unsigned int x = __float_as_uint(f);
  unsigned int r = x + 0x7FFFu + ((x >> 16) & 1u);
  return (unsigned short)(r >> 16);
}

union V16 { v16bf v; unsigned short u[16]; };

// ---------------------------------------------------------------------------
// Kernel 1: codebook f32 -> bf16, plus ||c||^2 per code (f32).
// ---------------------------------------------------------------------------
__global__ void vq_prep(const float* __restrict__ cb,
                        unsigned short* __restrict__ cb_bf,
                        float* __restrict__ cnorm) {
  __shared__ float red[256];
  const int row = blockIdx.x;
  const int t   = threadIdx.x;
  float v = cb[row * DIMS + t];
  cb_bf[row * DIMS + t] = f2bf(v);
  red[t] = v * v;
  __syncthreads();
  for (int s = 128; s > 0; s >>= 1) {
    if (t < s) red[t] += red[t + s];
    __syncthreads();
  }
  if (t == 0) cnorm[row] = red[0];
}

// ---------------------------------------------------------------------------
// Kernel 2: fused GEMM + argmin via V_WMMA_F32_16X16X32_BF16.
// 8 waves / block; each wave owns a 16-row tile (block covers 128 rows).
// A panel (16 x 256 bf16) held in registers; codebook tiles double-buffered
// in LDS (2 x 8 KB): tile nt+1 is DMA'd (async-to-LDS) while the 8 WMMAs of
// tile nt run. Two independent accumulator chains for XDL pipelining.
// ---------------------------------------------------------------------------
__global__ void __launch_bounds__(256) vq_argmin(
    const float* __restrict__ z_e,
    const unsigned short* __restrict__ cb_bf,
    const float* __restrict__ cnorm,
    int* __restrict__ idx_out,
    float* __restrict__ idxf_out) {
  __shared__ __align__(32) unsigned short tile[2][16 * DIMS];  // 16 KB

  const int t       = threadIdx.x;
  const int wave    = t >> 5;
  const int lane    = t & 31;
  const int half    = lane >> 4;        // 0: lanes 0-15, 1: lanes 16-31
  const int l15     = lane & 15;
  const int rowBase = blockIdx.x * 128 + wave * 16;

  // ---- Stage n-tile 0 (issue early so it overlaps the A-panel loads).
#if HAVE_ASYNC_LDS
  {
    const unsigned short* s = cb_bf + (size_t)(t >> 4) * DIMS + (t & 15) * 16;
    unsigned short* d = tile[0] + (t >> 4) * DIMS + (t & 15) * 16;
    async_cp16<0>(s, d);
    async_cp16<16>(s, d);
  }
#else
  uint4 p0, p1;
  {
    const uint4* s = (const uint4*)(cb_bf + (size_t)(t >> 4) * DIMS) + (t & 15) * 2;
    p0 = s[0]; p1 = s[1];
  }
#endif

  // ---- Load full A panel (16 rows x K=256, bf16) into 64 VGPRs/lane.
  // A-layout: per-lane 16 values = one contiguous 16-element K run
  // (low lanes: k 0..15 of the step; high lanes: k 16..31).
  V16 a[8];
  {
    const float* zp = z_e + (size_t)(rowBase + l15) * DIMS;
#pragma unroll
    for (int kt = 0; kt < 8; ++kt) {
      const int k0 = kt * 32 + half * 16;
      float f[16];
      *(float4*)(f + 0)  = *(const float4*)(zp + k0);
      *(float4*)(f + 4)  = *(const float4*)(zp + k0 + 4);
      *(float4*)(f + 8)  = *(const float4*)(zp + k0 + 8);
      *(float4*)(f + 12) = *(const float4*)(zp + k0 + 12);
#pragma unroll
      for (int j = 0; j < 16; ++j) a[kt].u[j] = f2bf(f[j]);
    }
  }

#if HAVE_ASYNC_LDS
  ASYNC_WAIT();
#else
  {
    uint4* d = (uint4*)(tile[0] + (t >> 4) * DIMS) + (t & 15) * 2;
    d[0] = p0; d[1] = p1;
  }
#endif
  __syncthreads();

  float best[8];
  int   bidx[8];
#pragma unroll
  for (int i = 0; i < 8; ++i) { best[i] = 3.402823466e38f; bidx[i] = 0; }

  for (int nt = 0; nt < KCODES / 16; ++nt) {
    const int  cur       = nt & 1;
    const int  nxt       = cur ^ 1;
    const bool have_next = (nt + 1 < KCODES / 16);

    // ---- Prefetch tile nt+1 into the other LDS buffer.
#if HAVE_ASYNC_LDS
    if (have_next) {
      const unsigned short* s =
          cb_bf + (size_t)((nt + 1) * 16 + (t >> 4)) * DIMS + (t & 15) * 16;
      unsigned short* d = tile[nxt] + (t >> 4) * DIMS + (t & 15) * 16;
      async_cp16<0>(s, d);
      async_cp16<16>(s, d);
    }
#else
    uint4 n0, n1;
    uint4* dstp = nullptr;
    if (have_next) {
      const uint4* s =
          (const uint4*)(cb_bf + (size_t)((nt + 1) * 16 + (t >> 4)) * DIMS)
          + (t & 15) * 2;
      n0 = s[0]; n1 = s[1];
      dstp = (uint4*)(tile[nxt] + (t >> 4) * DIMS) + (t & 15) * 2;
    }
#endif

    const float cn = cnorm[nt * 16 + l15];

    // ---- 8 WMMAs over K=256, two independent accumulator chains.
    const unsigned short* tbase = tile[cur] + l15 * DIMS + half * 16;
    v8f acc0 = {}, acc1 = {};
#pragma unroll
    for (int kt = 0; kt < 8; kt += 2) {
      const V16* b0 = (const V16*)(tbase + kt * 32);
      const V16* b1 = (const V16*)(tbase + (kt + 1) * 32);
      acc0 = __builtin_amdgcn_wmma_f32_16x16x32_bf16(
          false, a[kt].v, false, b0->v, (short)0, acc0, false, false);
      acc1 = __builtin_amdgcn_wmma_f32_16x16x32_bf16(
          false, a[kt + 1].v, false, b1->v, (short)0, acc1, false, false);
    }

#if !HAVE_ASYNC_LDS
    if (have_next) { dstp[0] = n0; dstp[1] = n1; }
#endif

    const int col = nt * 16 + l15;
#pragma unroll
    for (int i = 0; i < 8; ++i) {
      float s = acc0[i] + acc1[i];
      float d = fmaf(-2.0f, s, cn);            // ||c||^2 - 2 x.c
      if (d < best[i]) { best[i] = d; bidx[i] = col; }
    }

#if HAVE_ASYNC_LDS
    ASYNC_WAIT();
#endif
    __syncthreads();
  }

  // ---- Cross-lane argmin within each 16-lane half (C layout: VGPR i holds
  // row i for lanes 0-15, row i+8 for lanes 16-31). Tie-break: lower index.
#pragma unroll
  for (int i = 0; i < 8; ++i) {
#pragma unroll
    for (int m = 1; m <= 8; m <<= 1) {
      float ob = __shfl_xor(best[i], m, 32);
      int   oi = __shfl_xor(bidx[i], m, 32);
      if (ob < best[i] || (ob == best[i] && oi < bidx[i])) {
        best[i] = ob; bidx[i] = oi;
      }
    }
  }

  if (l15 == 0) {
#pragma unroll
    for (int i = 0; i < 8; ++i) {
      const int row = rowBase + half * 8 + i;
      idx_out[row]  = bidx[i];
      idxf_out[row] = (float)bidx[i];
    }
  }
}

// ---------------------------------------------------------------------------
// Kernel 3: gather z_q = codebook[idx] (exact f32) + per-block partial MSE.
// ---------------------------------------------------------------------------
__global__ void vq_gather(const float* __restrict__ z_e,
                          const float* __restrict__ cb,
                          const int* __restrict__ idx,
                          float* __restrict__ zq,
                          float* __restrict__ partial) {
  __shared__ float red[256];
  const int t = threadIdx.x;
  float lsum = 0.0f;
  const int base = blockIdx.x * 128;
  for (int r = 0; r < 128; ++r) {
    const int row = base + r;
    const int id  = idx[row];
    float c = cb[(size_t)id * DIMS + t];
    float x = z_e[(size_t)row * DIMS + t];
    zq[(size_t)row * DIMS + t] = c;
    float d = c - x;
    lsum = fmaf(d, d, lsum);
  }
  red[t] = lsum;
  __syncthreads();
  for (int s = 128; s > 0; s >>= 1) {
    if (t < s) red[t] += red[t + s];
    __syncthreads();
  }
  if (t == 0) partial[blockIdx.x] = red[0];
}

// ---------------------------------------------------------------------------
// Kernel 4: deterministic fixed-order final loss reduction.
// ---------------------------------------------------------------------------
__global__ void vq_finalize(const float* __restrict__ partial,
                            float* __restrict__ lossOut) {
  float s = 0.0f;
  for (int i = 0; i < 512; ++i) s += partial[i];
  *lossOut = s * (1.0f / (float)ZQ_ELEMS);
}

// ---------------------------------------------------------------------------
extern "C" void kernel_launch(void* const* d_in, const int* in_sizes, int n_in,
                              void* d_out, int out_size, void* d_ws, size_t ws_size,
                              hipStream_t stream) {
  const float* z_e = (const float*)d_in[0];
  const float* cb  = (const float*)d_in[1];
  float* out = (float*)d_out;
  char*  ws  = (char*)d_ws;

  // Workspace layout (~774 KB total):
  unsigned short* cb_bf = (unsigned short*)(ws);                      // 512 KB
  float* cnorm   = (float*)(ws + 512 * 1024);                        //   4 KB
  int*   idx     = (int*)(ws + 512 * 1024 + 4 * 1024);               // 256 KB
  float* partial = (float*)(ws + 512 * 1024 + 4 * 1024 + 256 * 1024);//   2 KB

  float* zq      = out;
  float* idxf    = out + (size_t)ZQ_ELEMS;
  float* lossOut = out + (size_t)ZQ_ELEMS + NROWS;

  vq_prep    <<<KCODES,      256, 0, stream>>>(cb, cb_bf, cnorm);
  vq_argmin  <<<NROWS / 128, 256, 0, stream>>>(z_e, cb_bf, cnorm, idx, idxf);
  vq_gather  <<<NROWS / 128, 256, 0, stream>>>(z_e, cb, idx, zq, partial);
  vq_finalize<<<1,           1,   0, stream>>>(partial, lossOut);
}
